// PatchAttention_83459804496250
// MI455X (gfx1250) — matrix operand order repaired and checked by reading, not compiled, not run on hardware
//
#include <hip/hip_runtime.h>
#include <hip/hip_bf16.h>
#include <cstdint>

typedef _Float16 f16;
typedef __attribute__((ext_vector_type(16))) _Float16 v16h;
typedef __attribute__((ext_vector_type(8)))  float    v8f;

#define HH 160
#define WW 160
#define HW 25600
#define NB 16
#define C3 192

// ---------------- weight f32 -> f16 conversion ----------------
__global__ void __launch_bounds__(256) k_wconv(const float* __restrict__ exp_w,
                                               const float* __restrict__ res_w,
                                               const float* __restrict__ fus_w,
                                               f16* __restrict__ expH,
                                               f16* __restrict__ resH,
                                               f16* __restrict__ fusH)
{
    int i = blockIdx.x * 256 + threadIdx.x;
    if (i < 192 * 64) expH[i] = (f16)exp_w[i];
    if (i < 64 * 192) resH[i] = (f16)res_w[i];
    if (i < 64 * 64)  fusH[i] = (f16)fus_w[i];
}

// ---------------- K1: ex = exp_w(192x64) @ x + exp_b  (WMMA) ----------------
// block: 256 thr (8 waves), tile: 192 out-ch x 64 pixels, K=64 (2 steps of 32)
__global__ void __launch_bounds__(256) k_ex(const float* __restrict__ x,
                                            const f16* __restrict__ expH,
                                            const float* __restrict__ exp_b,
                                            float* __restrict__ ex)
{
    __shared__ f16 Xh[64 * 64];
    const int n = blockIdx.y;
    const int pxbase = blockIdx.x * 64;
    const int tid = threadIdx.x;
    for (int idx = tid; idx < 64 * 64; idx += 256) {
        int k = idx >> 6, px = idx & 63;
        Xh[idx] = (f16)x[(n * 64 + k) * HW + pxbase + px];
    }
    __syncthreads();
    const int lane = tid & 31;
    const int wv = tid >> 5;
    const int m = lane & 15, hi = lane >> 4;
    for (int t6 = 0; t6 < 6; ++t6) {
        int tile = wv + 8 * t6;       // 48 tiles = 12 rt x 4 pt
        int rt = tile >> 2, pt = tile & 3;
        v8f acc = {0.f, 0.f, 0.f, 0.f, 0.f, 0.f, 0.f, 0.f};
#pragma unroll
        for (int kt = 0; kt < 2; ++kt) {
            v16h a, bf;
            int r = rt * 16 + m;
#pragma unroll
            for (int e = 0; e < 16; ++e) {       // A 16x32 f16 lane layout
                int K = kt * 32 + hi * 8 + ((e >> 3) << 4) + (e & 7);
                a[e] = expH[r * 64 + K];
            }
            int px = pt * 16 + m;
#pragma unroll
            for (int e = 0; e < 16; ++e) {       // B 32x16 f16 lane layout
                int K = kt * 32 + hi * 8 + ((e >> 3) << 4) + (e & 7);
                bf[e] = Xh[K * 64 + px];
            }
            acc = __builtin_amdgcn_wmma_f32_16x16x32_f16(false, a, false, bf,
                                                         (short)0, acc, false, false);
        }
#pragma unroll
        for (int v = 0; v < 8; ++v) {
            int rowg = rt * 16 + hi * 8 + v;
            int colp = pxbase + pt * 16 + m;
            ex[(n * C3 + rowg) * HW + colp] = acc[v] + exp_b[rowg];
        }
    }
}

// ---------------- per-window compile-time config ----------------
template <int WIN>
struct WinCfg {
    static constexpr int SH  = 160 / WIN;
    static constexpr int KH  = SH + 160 % WIN;  // == KW
    static constexpr int KK  = KH * KH;
    static constexpr int L   = WIN * WIN;
    static constexpr int CKK = 64 * KK;
    static constexpr int B   = NB * L;          // psa batch count
};

// gather one element of the (scrambled) psa input:  g in [0, NB*CKK*L)
template <int WIN>
__device__ __forceinline__ float ld_chunk(const float* __restrict__ ex, int cb, int g)
{
    using Cg = WinCfg<WIN>;
    int n_  = g / (Cg::CKK * Cg::L);
    int r   = g - n_ * (Cg::CKK * Cg::L);
    int ckk = r / Cg::L;
    int l   = r - ckk * Cg::L;
    int c   = ckk / Cg::KK;
    int ij  = ckk - c * Cg::KK;
    int i   = ij / Cg::KH;
    int j   = ij - i * Cg::KH;
    int lh  = l / WIN;
    int lw  = l - lh * WIN;
    int row = lh * Cg::SH + i;
    int col = lw * Cg::SH + j;
    return ex[(n_ * C3 + cb + c) * HW + row * WW + col];
}

// ---------------- per-batch reductions: softmax(q), xq, xmean ----------------
template <int WIN>
__global__ void __launch_bounds__(256) k_stats(const float* __restrict__ ex, int cb,
                                               const float* __restrict__ wq_w,
                                               const float* __restrict__ wq_b,
                                               float* __restrict__ qbuf,
                                               float* __restrict__ xq,
                                               float* __restrict__ xmean)
{
    using Cg = WinCfg<WIN>;
    __shared__ float sA[256], sB[256], swq[64];
    __shared__ float s_qmax, s_inv;
    const int b = blockIdx.x;
    const int tid = threadIdx.x;
    const int base = b * Cg::CKK;
    if (tid < 64) swq[tid] = wq_w[tid];
    __syncthreads();
    // phase 1: q per pixel + max
    float lmax = -3.4e38f;
    const float qb = wq_b[0];
    for (int p = tid; p < Cg::KK; p += 256) {
        float q = qb;
#pragma unroll 8
        for (int c = 0; c < 64; ++c)
            q += swq[c] * ld_chunk<WIN>(ex, cb, base + c * Cg::KK + p);
        qbuf[b * Cg::KK + p] = q;
        lmax = fmaxf(lmax, q);
    }
    sA[tid] = lmax; __syncthreads();
    for (int s = 128; s > 0; s >>= 1) { if (tid < s) sA[tid] = fmaxf(sA[tid], sA[tid + s]); __syncthreads(); }
    if (tid == 0) s_qmax = sA[0];
    __syncthreads();
    // phase 1b: exponentiate + sum
    float lsum = 0.f;
    const float qm = s_qmax;
    for (int p = tid; p < Cg::KK; p += 256) {
        float e = __expf(qbuf[b * Cg::KK + p] - qm);
        qbuf[b * Cg::KK + p] = e;
        lsum += e;
    }
    sA[tid] = lsum; __syncthreads();
    for (int s = 128; s > 0; s >>= 1) { if (tid < s) sA[tid] += sA[tid + s]; __syncthreads(); }
    if (tid == 0) s_inv = 1.0f / sA[0];
    __syncthreads();
    const float inv = s_inv;
    // phase 2: per-channel weighted/plain sums (fixed-order reductions -> deterministic)
    for (int c = 0; c < 64; ++c) {
        float sxq = 0.f, sxs = 0.f;
        for (int p = tid; p < Cg::KK; p += 256) {
            float v = ld_chunk<WIN>(ex, cb, base + c * Cg::KK + p);
            float e = qbuf[b * Cg::KK + p];
            sxq += v * e; sxs += v;
        }
        sA[tid] = sxq; sB[tid] = sxs; __syncthreads();
        for (int s = 128; s > 0; s >>= 1) {
            if (tid < s) { sA[tid] += sA[tid + s]; sB[tid] += sB[tid + s]; }
            __syncthreads();
        }
        if (tid == 0) {
            xq[b * 64 + c]    = sA[0] * inv;
            xmean[b * 64 + c] = sB[0] * (1.0f / Cg::KK);
        }
        __syncthreads();
    }
}

// ---------------- per-batch small math: ch_att, v_eff, bias_eff ----------------
template <int WIN>
__global__ void __launch_bounds__(64) k_patch(const float* __restrict__ xq,
                                              const float* __restrict__ xmean,
                                              const float* __restrict__ wv_w,
                                              const float* __restrict__ wv_b,
                                              const float* __restrict__ wz_w,
                                              const float* __restrict__ wz_b,
                                              const float* __restrict__ ln_g,
                                              const float* __restrict__ ln_b,
                                              const float* __restrict__ sp_wv_w,
                                              const float* __restrict__ sp_wv_b,
                                              const float* __restrict__ sp_wq_w,
                                              const float* __restrict__ sp_wq_b,
                                              float* __restrict__ ch_att,
                                              float* __restrict__ veff,
                                              float* __restrict__ be)
{
    __shared__ float sxq[64], sxm[64], swz[32], sz[64], sq[32], red[64];
    __shared__ float s_mu, s_rstd;
    const int b = blockIdx.x;
    const int t = threadIdx.x;
    sxq[t] = xq[b * 64 + t];
    sxm[t] = xmean[b * 64 + t];
    __syncthreads();
    if (t < 32) {
        float s = wv_b[t];
        for (int c = 0; c < 64; ++c) s += wv_w[t * 64 + c] * sxq[c];
        swz[t] = s;                                  // wz[k] = Wv . xq_norm + bv
        float u = sp_wq_b[t];
        for (int c = 0; c < 64; ++c) u += sp_wq_w[t * 64 + c] * sxm[c];
        sq[t] = u;                                   // pre-softmax spatial logits
    }
    __syncthreads();
    {
        float s = wz_b[t];
        for (int k = 0; k < 32; ++k) s += wz_w[t * 32 + k] * swz[k];
        sz[t] = s;                                   // z[o]
    }
    red[t] = sz[t]; __syncthreads();
    for (int s = 32; s > 0; s >>= 1) { if (t < s) red[t] += red[t + s]; __syncthreads(); }
    if (t == 0) s_mu = red[0] * (1.0f / 64.0f);
    __syncthreads();
    float d = sz[t] - s_mu;
    red[t] = d * d; __syncthreads();
    for (int s = 32; s > 0; s >>= 1) { if (t < s) red[t] += red[t + s]; __syncthreads(); }
    if (t == 0) s_rstd = rsqrtf(red[0] * (1.0f / 64.0f) + 1e-5f);
    __syncthreads();
    float zn = d * s_rstd * ln_g[t] + ln_b[t];
    ch_att[b * 64 + t] = 1.0f / (1.0f + __expf(-zn));
    // softmax over 32 spatial-query logits (serial in thread 0: only 32 elems)
    if (t == 0) {
        float mx = sq[0];
        for (int k = 1; k < 32; ++k) mx = fmaxf(mx, sq[k]);
        float ss = 0.f;
        for (int k = 0; k < 32; ++k) { float e = __expf(sq[k] - mx); sq[k] = e; ss += e; }
        float iv = 1.0f / ss;
        for (int k = 0; k < 32; ++k) sq[k] *= iv;
    }
    __syncthreads();
    float v = 0.f;
    for (int k = 0; k < 32; ++k) v += sq[k] * sp_wv_w[k * 64 + t];
    veff[b * 64 + t] = v;                            // v_eff = swq . sp_wv_w
    if (t == 0) {
        float s = 0.f;
        for (int k = 0; k < 32; ++k) s += sq[k] * sp_wv_b[k];
        be[b] = s;
    }
}

// ---------------- smap[b,p'] = sigmoid(v_eff[b] . X[b,:,p'] + be[b]) ----------------
template <int WIN>
__global__ void __launch_bounds__(256) k_smap(const float* __restrict__ ex, int cb,
                                              const float* __restrict__ veff,
                                              const float* __restrict__ be,
                                              float* __restrict__ smap)
{
    using Cg = WinCfg<WIN>;
    int idx = blockIdx.x * 256 + threadIdx.x;
    if (idx >= Cg::B * Cg::KK) return;
    int b = idx / Cg::KK;
    int p = idx - b * Cg::KK;
    int base = b * Cg::CKK + p;
    float s = be[b];
    const float* vb = veff + b * 64;
#pragma unroll 8
    for (int c = 0; c < 64; ++c)
        s += vb[c] * ld_chunk<WIN>(ex, cb, base + c * Cg::KK);
    smap[idx] = 1.0f / (1.0f + __expf(-s));
}

// ---------------- fold-average + add-ex, gather form (in place on ex -> cat) ----------------
template <int WIN>
__global__ void __launch_bounds__(256) k_apply(float* __restrict__ ex, int cb,
                                               const float* __restrict__ ch_att,
                                               const float* __restrict__ smap)
{
    using Cg = WinCfg<WIN>;
    int idx = blockIdx.x * 256 + threadIdx.x;   // n*HW + pp, exact grid
    int n = idx / HW;
    int pp = idx - n * HW;
    int row = pp / WW, col = pp - row * WW;
    int lh0 = row - Cg::KH + Cg::SH; lh0 = lh0 > 0 ? lh0 / Cg::SH : 0;
    int lh1 = row / Cg::SH; if (lh1 > WIN - 1) lh1 = WIN - 1;
    int lw0 = col - Cg::KH + Cg::SH; lw0 = lw0 > 0 ? lw0 / Cg::SH : 0;
    int lw1 = col / Cg::SH; if (lw1 > WIN - 1) lw1 = WIN - 1;
    float icnt = 1.0f / (float)((lh1 - lh0 + 1) * (lw1 - lw0 + 1));
    for (int c = 0; c < 64; ++c) {
        int a = (n * C3 + cb + c) * HW + pp;
        float exv = ex[a];
        float acc = 0.f;
        for (int lh = lh0; lh <= lh1; ++lh)
            for (int lw = lw0; lw <= lw1; ++lw) {
                int i = row - lh * Cg::SH, j = col - lw * Cg::SH;
                int l = lh * WIN + lw;
                int g = ((n * 64 + c) * Cg::KK + i * Cg::KH + j) * Cg::L + l;
                int b = g / Cg::CKK;
                int rem = g - b * Cg::CKK;
                int c2 = rem / Cg::KK;
                int p2 = rem - c2 * Cg::KK;
                acc += ch_att[b * 64 + c2] + smap[b * Cg::KK + p2];
            }
        ex[a] = exv * (1.0f + acc * icnt);      // cat = psa_out_avg + ex
    }
}

// ---------------- fused: out = fus_w @ (res_w @ cat + res_b + x) + fus_b (2x WMMA) ----------------
__global__ void __launch_bounds__(256) k_resfus(const float* __restrict__ cat,
                                                const float* __restrict__ x,
                                                const f16* __restrict__ resH,
                                                const float* __restrict__ res_b,
                                                const f16* __restrict__ fusH,
                                                const float* __restrict__ fus_b,
                                                float* __restrict__ out)
{
    __shared__ f16 CatH[192 * 64];
    __shared__ f16 Y1h[64 * 64];
    const int n = blockIdx.y;
    const int pxbase = blockIdx.x * 64;
    const int tid = threadIdx.x;
    for (int idx = tid; idx < 192 * 64; idx += 256) {
        int k = idx >> 6, px = idx & 63;
        CatH[idx] = (f16)cat[(n * C3 + k) * HW + pxbase + px];
    }
    __syncthreads();
    const int lane = tid & 31;
    const int wv = tid >> 5;
    const int m = lane & 15, hi = lane >> 4;
    // GEMM1: 64x64px, K=192
    for (int t2 = 0; t2 < 2; ++t2) {
        int tile = wv + 8 * t2;           // 16 tiles = 4 rt x 4 pt
        int rt = tile >> 2, pt = tile & 3;
        v8f acc = {0.f, 0.f, 0.f, 0.f, 0.f, 0.f, 0.f, 0.f};
        for (int kt = 0; kt < 6; ++kt) {
            v16h a, bf;
            int r = rt * 16 + m;
#pragma unroll
            for (int e = 0; e < 16; ++e) {
                int K = kt * 32 + hi * 8 + ((e >> 3) << 4) + (e & 7);
                a[e] = resH[r * 192 + K];
            }
            int px = pt * 16 + m;
#pragma unroll
            for (int e = 0; e < 16; ++e) {
                int K = kt * 32 + hi * 8 + ((e >> 3) << 4) + (e & 7);
                bf[e] = CatH[K * 64 + px];
            }
            acc = __builtin_amdgcn_wmma_f32_16x16x32_f16(false, a, false, bf,
                                                         (short)0, acc, false, false);
        }
#pragma unroll
        for (int v = 0; v < 8; ++v) {
            int rowg = rt * 16 + hi * 8 + v;
            int colp = pt * 16 + m;
            float y1 = acc[v] + res_b[rowg] + x[(n * 64 + rowg) * HW + pxbase + colp];
            Y1h[rowg * 64 + colp] = (f16)y1;
        }
    }
    __syncthreads();
    // GEMM2: 64x64px, K=64
    for (int t2 = 0; t2 < 2; ++t2) {
        int tile = wv + 8 * t2;
        int rt = tile >> 2, pt = tile & 3;
        v8f acc = {0.f, 0.f, 0.f, 0.f, 0.f, 0.f, 0.f, 0.f};
#pragma unroll
        for (int kt = 0; kt < 2; ++kt) {
            v16h a, bf;
            int r = rt * 16 + m;
#pragma unroll
            for (int e = 0; e < 16; ++e) {
                int K = kt * 32 + hi * 8 + ((e >> 3) << 4) + (e & 7);
                a[e] = fusH[r * 64 + K];
            }
            int px = pt * 16 + m;
#pragma unroll
            for (int e = 0; e < 16; ++e) {
                int K = kt * 32 + hi * 8 + ((e >> 3) << 4) + (e & 7);
                bf[e] = Y1h[K * 64 + px];
            }
            acc = __builtin_amdgcn_wmma_f32_16x16x32_f16(false, a, false, bf,
                                                         (short)0, acc, false, false);
        }
#pragma unroll
        for (int v = 0; v < 8; ++v) {
            int rowg = rt * 16 + hi * 8 + v;
            int colp = pxbase + pt * 16 + m;
            out[(n * 64 + rowg) * HW + colp] = acc[v] + fus_b[rowg];
        }
    }
}

// ---------------- host-side per-window dispatch ----------------
template <int WIN>
static void run_window(float* ex, int cb,
                       const float* ch_wv_w, const float* ch_wv_b,
                       const float* ch_wq_w, const float* ch_wq_b,
                       const float* ch_wz_w, const float* ch_wz_b,
                       const float* ln_g, const float* ln_b,
                       const float* sp_wv_w, const float* sp_wv_b,
                       const float* sp_wq_w, const float* sp_wq_b,
                       float* qbuf, float* xq, float* xmean,
                       float* chat, float* veff, float* be,
                       float* smap, hipStream_t stream)
{
    using Cg = WinCfg<WIN>;
    k_stats<WIN><<<Cg::B, 256, 0, stream>>>(ex, cb, ch_wq_w, ch_wq_b, qbuf, xq, xmean);
    k_patch<WIN><<<Cg::B, 64, 0, stream>>>(xq, xmean, ch_wv_w, ch_wv_b, ch_wz_w, ch_wz_b,
                                           ln_g, ln_b, sp_wv_w, sp_wv_b, sp_wq_w, sp_wq_b,
                                           chat, veff, be);
    int tot = Cg::B * Cg::KK;
    k_smap<WIN><<<(tot + 255) / 256, 256, 0, stream>>>(ex, cb, veff, be, smap);
    k_apply<WIN><<<(NB * HW) / 256, 256, 0, stream>>>(ex, cb, chat, smap);
}

extern "C" void kernel_launch(void* const* d_in, const int* in_sizes, int n_in,
                              void* d_out, int out_size, void* d_ws, size_t ws_size,
                              hipStream_t stream)
{
    const float* x       = (const float*)d_in[0];
    const float* exp_w   = (const float*)d_in[1];
    const float* exp_b   = (const float*)d_in[2];
    const float* res_w   = (const float*)d_in[3];
    const float* res_b   = (const float*)d_in[4];
    const float* fus_w   = (const float*)d_in[5];
    const float* fus_b   = (const float*)d_in[6];
    const float* ch_wv_w = (const float*)d_in[7];
    const float* ch_wv_b = (const float*)d_in[8];
    const float* ch_wq_w = (const float*)d_in[9];
    const float* ch_wq_b = (const float*)d_in[10];
    const float* ch_wz_w = (const float*)d_in[11];
    const float* ch_wz_b = (const float*)d_in[12];
    const float* ln_g    = (const float*)d_in[13];
    const float* ln_b    = (const float*)d_in[14];
    const float* sp_wv_w = (const float*)d_in[15];
    const float* sp_wv_b = (const float*)d_in[16];
    const float* sp_wq_w = (const float*)d_in[17];
    const float* sp_wq_b = (const float*)d_in[18];
    float* out = (float*)d_out;

    char* ws = (char*)d_ws;
    size_t off = 0;
    auto alloc = [&](size_t bytes) {
        off = (off + 255) & ~(size_t)255;
        void* p = ws + off;
        off += bytes;
        return p;
    };
    float* ex    = (float*)alloc((size_t)NB * C3 * HW * 4);  // ex, becomes cat in place
    float* qbuf  = (float*)alloc(419904 * 4);                // max B*KK over windows
    float* smap  = (float*)alloc(419904 * 4);
    float* xq    = (float*)alloc(400 * 64 * 4);
    float* xmean = (float*)alloc(400 * 64 * 4);
    float* chat  = (float*)alloc(400 * 64 * 4);
    float* veff  = (float*)alloc(400 * 64 * 4);
    float* be    = (float*)alloc(400 * 4);
    f16* expH    = (f16*)alloc(192 * 64 * 2);
    f16* resH    = (f16*)alloc(64 * 192 * 2);
    f16* fusH    = (f16*)alloc(64 * 64 * 2);

    k_wconv<<<48, 256, 0, stream>>>(exp_w, res_w, fus_w, expH, resH, fusH);
    k_ex<<<dim3(HW / 64, NB), 256, 0, stream>>>(x, expH, exp_b, ex);

    run_window<1>(ex, 0,   ch_wv_w, ch_wv_b, ch_wq_w, ch_wq_b, ch_wz_w, ch_wz_b,
                  ln_g, ln_b, sp_wv_w, sp_wv_b, sp_wq_w, sp_wq_b,
                  qbuf, xq, xmean, chat, veff, be, smap, stream);
    run_window<3>(ex, 64,  ch_wv_w, ch_wv_b, ch_wq_w, ch_wq_b, ch_wz_w, ch_wz_b,
                  ln_g, ln_b, sp_wv_w, sp_wv_b, sp_wq_w, sp_wq_b,
                  qbuf, xq, xmean, chat, veff, be, smap, stream);
    run_window<5>(ex, 128, ch_wv_w, ch_wv_b, ch_wq_w, ch_wq_b, ch_wz_w, ch_wz_b,
                  ln_g, ln_b, sp_wv_w, sp_wv_b, sp_wq_w, sp_wq_b,
                  qbuf, xq, xmean, chat, veff, be, smap, stream);

    k_resfus<<<dim3(HW / 64, NB), 256, 0, stream>>>(ex, x, resH, res_b, fusH, fus_b, out);
}
